// HANLayer_29059748725073
// MI455X (gfx1250) — compile-verified, matrix operand order checked by reading, not run on hardware
//
#include <hip/hip_runtime.h>
#include <hip/hip_bf16.h>

#define NN    50000
#define NE    800000
#define MP    3
#define IND   128
#define DF    256      /* HEADS*OUT_DIM */
#define NHEAD 8
#define OD    32
#define HIDD  128
#define SLOPE 0.2f

typedef __attribute__((ext_vector_type(16))) __bf16   v16bf;
typedef __attribute__((ext_vector_type(8)))  float    v8f;
typedef __attribute__((ext_vector_type(4)))  unsigned u32x4;

union ABPack { u32x4 u[2]; v16bf v; };

__device__ __forceinline__ __bf16 f2bf(float f) {
    unsigned u = __float_as_uint(f);
    unsigned r = (u + 0x7fffu + ((u >> 16) & 1u)) >> 16;   // round-to-nearest-even
    unsigned short s = (unsigned short)r;
    __bf16 b;
    __builtin_memcpy(&b, &s, 2);
    return b;
}

// order-preserving float -> u32 key for atomicMax-based segment max
__device__ __forceinline__ unsigned fkey(float f) {
    unsigned u = __float_as_uint(f);
    return (u & 0x80000000u) ? ~u : (u | 0x80000000u);
}
__device__ __forceinline__ float fdec(unsigned u) {
    unsigned i = (u & 0x80000000u) ? (u & 0x7fffffffu) : ~u;
    return __uint_as_float(i);
}

__device__ __forceinline__ float lrelu(float v) { return v > 0.f ? v : SLOPE * v; }

// ---------------- prep: f32 -> bf16 conversions / transposes ----------------

__global__ void k_cvt_h(const float* __restrict__ h, __bf16* __restrict__ Abf) {
    int i = blockIdx.x * blockDim.x + threadIdx.x;
    if (i < NN * IND) Abf[i] = f2bf(h[i]);
}

// W_gat (MP,IND,DF) -> Wt (MP,DF,IND)  (B columns become contiguous rows)
__global__ void k_cvt_wt(const float* __restrict__ W, __bf16* __restrict__ Wt) {
    int i = blockIdx.x * blockDim.x + threadIdx.x;
    if (i >= MP * IND * DF) return;
    int m = i / (IND * DF), r = i % (IND * DF);
    int k = r / DF, n = r % DF;
    Wt[(size_t)m * DF * IND + (size_t)n * IND + k] = f2bf(W[i]);
}

// W1 (DF,HIDD) -> W1t (HIDD,DF)
__global__ void k_cvt_w1t(const float* __restrict__ W1, __bf16* __restrict__ W1t) {
    int i = blockIdx.x * blockDim.x + threadIdx.x;
    if (i >= HIDD * DF) return;
    int j = i / DF, k = i % DF;
    W1t[(size_t)j * DF + k] = f2bf(W1[(size_t)k * HIDD + j]);
}

__global__ void k_fill_u32(unsigned* __restrict__ p, unsigned val, long long n) {
    long long i = (long long)blockIdx.x * blockDim.x + threadIdx.x;
    long long s = (long long)gridDim.x * blockDim.x;
    for (; i < n; i += s) p[i] = val;
}

// ---------------- GEMM 1: feat = h @ W_gat[m]  (WMMA bf16 -> f32) -----------

__global__ __launch_bounds__(256) void k_gemm_feat(const __bf16* __restrict__ Abf,
                                                   const __bf16* __restrict__ Bt,  // DF x IND
                                                   float* __restrict__ feat) {
    int wave = (blockIdx.x * blockDim.x + threadIdx.x) >> 5;
    if (wave >= (NN / 16) * (DF / 16)) return;
    int rowTile = wave / (DF / 16), colTile = wave % (DF / 16);
    int lane = threadIdx.x & 31, r = lane & 15, half = lane >> 4;

    const __bf16* arow = Abf + (size_t)(rowTile * 16 + r) * IND;
    const __bf16* brow = Bt  + (size_t)(colTile * 16 + r) * IND;

    v8f acc = {};
#pragma unroll
    for (int kb = 0; kb < IND; kb += 32) {
        ABPack a, b;
        a.u[0] = *(const u32x4*)(arow + kb + 8 * half);
        a.u[1] = *(const u32x4*)(arow + kb + 16 + 8 * half);
        b.u[0] = *(const u32x4*)(brow + kb + 8 * half);
        b.u[1] = *(const u32x4*)(brow + kb + 16 + 8 * half);
        acc = __builtin_amdgcn_wmma_f32_16x16x32_bf16(false, a.v, false, b.v,
                                                      (short)0, acc, false, false);
    }
    int col = colTile * 16 + r;
#pragma unroll
    for (int v = 0; v < 8; ++v)
        feat[(size_t)(rowTile * 16 + v + 8 * half) * DF + col] = acc[v];
}

// ---------------- attention logits per node ---------------------------------

__global__ void k_el_er(const float* __restrict__ feat,
                        const float* __restrict__ al, const float* __restrict__ ar,
                        float* __restrict__ el, float* __restrict__ er) {
    int idx = blockIdx.x * blockDim.x + threadIdx.x;   // n*NHEAD + k
    if (idx >= NN * NHEAD) return;
    int n = idx >> 3, k = idx & 7;
    const float* f = feat + (size_t)n * DF + k * OD;
    const float* L = al + k * OD;
    const float* R = ar + k * OD;
    float sl = 0.f, sr = 0.f;
#pragma unroll
    for (int d = 0; d < OD; ++d) { sl += f[d] * L[d]; sr += f[d] * R[d]; }
    el[idx] = sl; er[idx] = sr;
}

// ---------------- edge softmax (3 passes) -----------------------------------

__global__ void k_edge_max(const int* __restrict__ src, const int* __restrict__ dst,
                           const float* __restrict__ el, const float* __restrict__ er,
                           unsigned* __restrict__ emax) {
    for (int idx = blockIdx.x * blockDim.x + threadIdx.x; idx < NE * NHEAD;
         idx += gridDim.x * blockDim.x) {
        int e = idx >> 3, k = idx & 7;
        int s = src[e], t = dst[e];
        float v = lrelu(el[s * NHEAD + k] + er[t * NHEAD + k]);
        atomicMax(&emax[t * NHEAD + k], fkey(v));
    }
}

__global__ void k_edge_expsum(const int* __restrict__ src, const int* __restrict__ dst,
                              const float* __restrict__ el, const float* __restrict__ er,
                              const unsigned* __restrict__ emax, float* __restrict__ denom) {
    for (int idx = blockIdx.x * blockDim.x + threadIdx.x; idx < NE * NHEAD;
         idx += gridDim.x * blockDim.x) {
        int e = idx >> 3, k = idx & 7;
        int s = src[e], t = dst[e];
        float v = lrelu(el[s * NHEAD + k] + er[t * NHEAD + k]);
        atomicAdd(&denom[t * NHEAD + k], expf(v - fdec(emax[t * NHEAD + k])));
    }
}

// one wave per edge: lane = feature d, loop over 8 heads -> coalesced 128B ops
__global__ void k_edge_scatter(const int* __restrict__ src, const int* __restrict__ dst,
                               const float* __restrict__ el, const float* __restrict__ er,
                               const unsigned* __restrict__ emax, const float* __restrict__ denom,
                               const float* __restrict__ feat, float* __restrict__ zacc) {
    int gw = (blockIdx.x * blockDim.x + threadIdx.x) >> 5;
    int nw = (gridDim.x * blockDim.x) >> 5;
    int lane = threadIdx.x & 31;
    for (int e = gw; e < NE; e += nw) {
        int s = src[e], t = dst[e];
        const float* fs = feat + (size_t)s * DF;
        float* zt = zacc + (size_t)t * DF;
#pragma unroll
        for (int k = 0; k < NHEAD; ++k) {
            float v = lrelu(el[s * NHEAD + k] + er[t * NHEAD + k]);
            float alpha = expf(v - fdec(emax[t * NHEAD + k])) /
                          (denom[t * NHEAD + k] + 1e-9f);
            atomicAdd(&zt[k * OD + lane], alpha * fs[k * OD + lane]);
        }
    }
}

// z = elu(acc + bias); also emit bf16 copy for the semantic-attention GEMM
__global__ void k_finalize(float* __restrict__ zm, const float* __restrict__ bias,
                           __bf16* __restrict__ zbfm) {
    int idx = blockIdx.x * blockDim.x + threadIdx.x;
    if (idx >= NN * DF) return;
    int c = idx & (DF - 1);
    float v = zm[idx] + bias[c];
    v = v > 0.f ? v : expm1f(v);
    zm[idx] = v;
    zbfm[idx] = f2bf(v);
}

// ---------------- GEMM 2: semantic attention scores (WMMA) ------------------
// wsum[m] += sum_{rows,cols} w2[c] * tanh( (z @ W1)[r,c] + b1[c] )

__global__ __launch_bounds__(256) void k_sem_gemm(const __bf16* __restrict__ zbf,  // (MP*NN) x DF
                                                  const __bf16* __restrict__ W1t,  // HIDD x DF
                                                  const float* __restrict__ b1,
                                                  const float* __restrict__ w2,
                                                  float* __restrict__ wsum) {
    int wave = (blockIdx.x * blockDim.x + threadIdx.x) >> 5;
    const int nRowTiles = (MP * NN) / 16;   // 9375
    if (wave >= nRowTiles) return;
    int lane = threadIdx.x & 31, r = lane & 15, half = lane >> 4;
    int m = wave / (NN / 16);

    const __bf16* arow = zbf + (size_t)(wave * 16 + r) * DF;
    float partial = 0.f;

    for (int ct = 0; ct < HIDD / 16; ++ct) {
        const __bf16* brow = W1t + (size_t)(ct * 16 + r) * DF;
        v8f acc = {};
#pragma unroll
        for (int kb = 0; kb < DF; kb += 32) {
            ABPack a, b;
            a.u[0] = *(const u32x4*)(arow + kb + 8 * half);
            a.u[1] = *(const u32x4*)(arow + kb + 16 + 8 * half);
            b.u[0] = *(const u32x4*)(brow + kb + 8 * half);
            b.u[1] = *(const u32x4*)(brow + kb + 16 + 8 * half);
            acc = __builtin_amdgcn_wmma_f32_16x16x32_bf16(false, a.v, false, b.v,
                                                          (short)0, acc, false, false);
        }
        int col = ct * 16 + r;
        float wc = w2[col], bc = b1[col];
#pragma unroll
        for (int v = 0; v < 8; ++v) partial += wc * tanhf(acc[v] + bc);
    }
    for (int off = 16; off; off >>= 1) partial += __shfl_xor(partial, off, 32);
    if (lane == 0) atomicAdd(&wsum[m], partial);
}

__global__ void k_beta(const float* __restrict__ wsum, float* __restrict__ beta) {
    if (threadIdx.x != 0 || blockIdx.x != 0) return;
    float w0 = wsum[0] / (float)NN, w1 = wsum[1] / (float)NN, w2v = wsum[2] / (float)NN;
    float mx = fmaxf(w0, fmaxf(w1, w2v));
    float e0 = expf(w0 - mx), e1 = expf(w1 - mx), e2 = expf(w2v - mx);
    float s = e0 + e1 + e2;
    beta[0] = e0 / s; beta[1] = e1 / s; beta[2] = e2 / s;
}

__global__ void k_combine(const float* __restrict__ z, const float* __restrict__ beta,
                          float* __restrict__ out) {
    int idx = blockIdx.x * blockDim.x + threadIdx.x;
    if (idx >= NN * DF) return;
    out[idx] = beta[0] * z[idx] +
               beta[1] * z[(size_t)NN * DF + idx] +
               beta[2] * z[(size_t)2 * NN * DF + idx];
}

// ---------------- host launcher ---------------------------------------------

extern "C" void kernel_launch(void* const* d_in, const int* in_sizes, int n_in,
                              void* d_out, int out_size, void* d_ws, size_t ws_size,
                              hipStream_t stream) {
    (void)in_sizes; (void)n_in; (void)out_size; (void)ws_size;
    const float* h      = (const float*)d_in[0];
    const float* W_gat  = (const float*)d_in[1];
    const float* attn_l = (const float*)d_in[2];
    const float* attn_r = (const float*)d_in[3];
    const float* bias_g = (const float*)d_in[4];
    const float* W1     = (const float*)d_in[5];
    const float* b1     = (const float*)d_in[6];
    const float* w2     = (const float*)d_in[7];
    const int*   src    = (const int*)d_in[8];
    const int*   dst    = (const int*)d_in[9];
    float* out = (float*)d_out;

    char* base = (char*)d_ws;
    size_t o = 0;
    auto alloc = [&](size_t bytes) -> char* {
        char* p = base + o;
        o += (bytes + 255) & ~(size_t)255;
        return p;
    };
    __bf16*  Abf   = (__bf16*)alloc((size_t)NN * IND * 2);
    __bf16*  Wt    = (__bf16*)alloc((size_t)MP * DF * IND * 2);
    __bf16*  W1t   = (__bf16*)alloc((size_t)HIDD * DF * 2);
    float*   feat  = (float*)alloc((size_t)NN * DF * 4);
    float*   el    = (float*)alloc((size_t)NN * NHEAD * 4);
    float*   er    = (float*)alloc((size_t)NN * NHEAD * 4);
    unsigned* emax = (unsigned*)alloc((size_t)NN * NHEAD * 4);
    float*   denom = (float*)alloc((size_t)NN * NHEAD * 4);
    float*   z     = (float*)alloc((size_t)MP * NN * DF * 4);
    __bf16*  zbf   = (__bf16*)alloc((size_t)MP * NN * DF * 2);
    float*   wsum  = (float*)alloc(256);
    float*   beta  = (float*)alloc(256);

    k_cvt_h  <<<(NN * IND + 255) / 256, 256, 0, stream>>>(h, Abf);
    k_cvt_wt <<<(MP * IND * DF + 255) / 256, 256, 0, stream>>>(W_gat, Wt);
    k_cvt_w1t<<<(HIDD * DF + 255) / 256, 256, 0, stream>>>(W1, W1t);

    for (int m = 0; m < MP; ++m) {
        const int* sm = src + (size_t)m * NE;
        const int* tm = dst + (size_t)m * NE;
        float* zm = z + (size_t)m * NN * DF;
        __bf16* zbfm = zbf + (size_t)m * NN * DF;

        k_gemm_feat<<<(NN / 16) * (DF / 16) / 8, 256, 0, stream>>>(
            Abf, Wt + (size_t)m * DF * IND, feat);
        k_el_er<<<(NN * NHEAD + 255) / 256, 256, 0, stream>>>(
            feat, attn_l + m * DF, attn_r + m * DF, el, er);
        k_fill_u32<<<1024, 256, 0, stream>>>(emax, 0u, (long long)NN * NHEAD);
        k_fill_u32<<<1024, 256, 0, stream>>>((unsigned*)denom, 0u, (long long)NN * NHEAD);
        k_edge_max<<<8192, 256, 0, stream>>>(sm, tm, el, er, emax);
        k_edge_expsum<<<8192, 256, 0, stream>>>(sm, tm, el, er, emax, denom);
        k_fill_u32<<<8192, 256, 0, stream>>>((unsigned*)zm, 0u, (long long)NN * DF);
        k_edge_scatter<<<4096, 256, 0, stream>>>(sm, tm, el, er, emax, denom, feat, zm);
        k_finalize<<<(NN * DF + 255) / 256, 256, 0, stream>>>(zm, bias_g + m * DF, zbfm);
    }

    k_fill_u32<<<1, 64, 0, stream>>>((unsigned*)wsum, 0u, 8);
    k_sem_gemm<<<((MP * NN / 16) + 7) / 8, 256, 0, stream>>>(zbf, W1t, b1, w2, wsum);
    k_beta<<<1, 1, 0, stream>>>(wsum, beta);
    k_combine<<<(NN * DF + 255) / 256, 256, 0, stream>>>(z, beta, out);
}
